// CausalSelfAttention_87771951661549
// MI455X (gfx1250) — matrix operand order repaired and checked by reading, not compiled, not run on hardware
//
#include <hip/hip_runtime.h>

#define Bz 4
#define Tz 2048
#define Ez 1024
#define Hz 16
#define Dz 64
// DH^-0.5 * log2(e): fold softmax scale + natural->base2 exp into Q at store time
#define QSCALE 0.1803368801111204f

typedef __attribute__((ext_vector_type(16))) __bf16        v16bf;
typedef __attribute__((ext_vector_type(8)))  float         v8f;
typedef __attribute__((ext_vector_type(4)))  unsigned int  v4u;

union Frag {
  v16bf          bf;
  v4u            u4[2];
  unsigned short us[16];
};

__device__ __forceinline__ unsigned short f2bf(float f) {
  unsigned int u = __float_as_uint(f);
  u += 0x7FFFu + ((u >> 16) & 1u);           // round-to-nearest-even
  return (unsigned short)(u >> 16);
}

__device__ __forceinline__ v8f wmma_bf16(v16bf a, v16bf b, v8f c) {
  return __builtin_amdgcn_wmma_f32_16x16x32_bf16(
      /*neg_a=*/false, a, /*neg_b=*/false, b,
      /*c_mod=*/(short)0, c, /*reuse_a=*/false, /*reuse_b=*/false);
}

// A-fragment (16x32 bf16, M=lane&15). Row-major source, stride in elements.
// lane<16 holds K=k0..k0+15; lane>=16 holds K=k0+16..k0+31.
__device__ __forceinline__ v16bf ldfragA(const unsigned short* p, int stride,
                                         int row, int k0, int hi) {
  const unsigned short* q = p + (size_t)row * stride + k0 + 16 * hi;
  Frag f;
  f.u4[0] = *(const v4u*)(q);
  f.u4[1] = *(const v4u*)(q + 8);
  return f.bf;
}

// B-fragment (32x16 bf16, N=lane&15) from an NxK row-major source
// (i.e. B[k][n] = src[n*stride + k]).  lane<16: K=k0..k0+15; lane>=16: +16.
__device__ __forceinline__ v16bf ldfragB(const unsigned short* p, int stride,
                                         int row, int k0, int hi) {
  const unsigned short* q = p + (size_t)row * stride + k0 + 16 * hi;
  Frag f;
  f.u4[0] = *(const v4u*)(q);
  f.u4[1] = *(const v4u*)(q + 8);
  return f.bf;
}

// -------------------------------------------------------------------------
// fp32 -> bf16 bulk conversion (all element counts are multiples of 4)
// -------------------------------------------------------------------------
__global__ void __launch_bounds__(256)
cvt_kernel(const float* __restrict__ in, unsigned short* __restrict__ out, int n) {
  int i = (blockIdx.x * 256 + threadIdx.x) * 4;
  if (i + 3 < n) {
    float4 f = *(const float4*)(in + i);
    union { unsigned short h[4]; uint2 u; } pk;
    pk.h[0] = f2bf(f.x); pk.h[1] = f2bf(f.y);
    pk.h[2] = f2bf(f.z); pk.h[3] = f2bf(f.w);
    *(uint2*)(out + i) = pk.u;
  }
}

// -------------------------------------------------------------------------
// GEMM core: C(32x64 per wave) = A(row-major MxK) * Bw(row-major NxK)^T
// 2 A-frags + 4 B-frags -> 8 WMMAs per K-step (1.5 b128 loads per WMMA).
// -------------------------------------------------------------------------
__device__ __forceinline__ void gemm_core(const unsigned short* __restrict__ A,
                                          const unsigned short* __restrict__ Bw,
                                          int K, int mbase, int nbase,
                                          int ln, int hi, v8f acc[2][4]) {
  for (int k0 = 0; k0 < K; k0 += 32) {
    v16bf a0 = ldfragA(A, K, mbase + ln,      k0, hi);
    v16bf a1 = ldfragA(A, K, mbase + 16 + ln, k0, hi);
    v16bf b0 = ldfragB(Bw, K, nbase + ln,      k0, hi);
    v16bf b1 = ldfragB(Bw, K, nbase + 16 + ln, k0, hi);
    v16bf b2 = ldfragB(Bw, K, nbase + 32 + ln, k0, hi);
    v16bf b3 = ldfragB(Bw, K, nbase + 48 + ln, k0, hi);
    acc[0][0] = wmma_bf16(a0, b0, acc[0][0]);
    acc[0][1] = wmma_bf16(a0, b1, acc[0][1]);
    acc[0][2] = wmma_bf16(a0, b2, acc[0][2]);
    acc[0][3] = wmma_bf16(a0, b3, acc[0][3]);
    acc[1][0] = wmma_bf16(a1, b0, acc[1][0]);
    acc[1][1] = wmma_bf16(a1, b1, acc[1][1]);
    acc[1][2] = wmma_bf16(a1, b2, acc[1][2]);
    acc[1][3] = wmma_bf16(a1, b3, acc[1][3]);
  }
}

// -------------------------------------------------------------------------
// QKV projection: qkv = x @ w_qkv^T + b_qkv, scattered to per-head buffers.
// Q gets QSCALE folded in; V is stored transposed [b,h,d,t].
// block = 256 threads (8 waves, 4m x 2n), block tile 128(M) x 128(N)
// -------------------------------------------------------------------------
__global__ void __launch_bounds__(256)
qkv_kernel(const unsigned short* __restrict__ xb,
           const unsigned short* __restrict__ wb,
           const float* __restrict__ bias,
           unsigned short* __restrict__ qO,
           unsigned short* __restrict__ kO,
           unsigned short* __restrict__ vtO) {
  const int lane = threadIdx.x & 31, wave = threadIdx.x >> 5;
  const int ln = lane & 15, hi = lane >> 4;
  const int wm = wave >> 1, wn = wave & 1;
  const int mbase = blockIdx.y * 128 + wm * 32;
  const int nbase = blockIdx.x * 128 + wn * 64;

  v8f acc[2][4] = {};
  gemm_core(xb, wb, Ez, mbase, nbase, ln, hi, acc);

  for (int nt = 0; nt < 4; ++nt) {
    const int n = nbase + nt * 16 + ln;
    const float bv = bias[n];
    const int which = n >> 10;           // 0=q, 1=k, 2=v
    const int wi = n & (Ez - 1);
    const int h = wi >> 6, d = wi & (Dz - 1);
    for (int mt = 0; mt < 2; ++mt) {
      for (int r = 0; r < 8; ++r) {
        const int m = mbase + mt * 16 + r + 8 * hi;
        const int b = m >> 11, t = m & (Tz - 1);
        const size_t bh = (size_t)(b * Hz + h);
        const float v = acc[mt][nt][r] + bv;
        if (which == 0)      qO[(bh * Tz + t) * Dz + d]  = f2bf(v * QSCALE);
        else if (which == 1) kO[(bh * Tz + t) * Dz + d]  = f2bf(v);
        else                 vtO[(bh * Dz + d) * Tz + t] = f2bf(v);
      }
    }
  }
}

// -------------------------------------------------------------------------
// Causal flash attention. One wave owns 16 queries; computes S^T = K*Q^T and
// O^T = V^T*P^T so the WMMA C-layout of S^T feeds the next WMMA B-operand
// with a single half-wave shuffle. Online softmax in exp2 domain.
// block = 128 threads (4 waves); grid = (T/64, B*H)
// -------------------------------------------------------------------------
__global__ void __launch_bounds__(128)
attn_kernel(const unsigned short* __restrict__ qB,
            const unsigned short* __restrict__ kB,
            const unsigned short* __restrict__ vtB,
            unsigned short* __restrict__ oB) {
  const int lane = threadIdx.x & 31, wave = threadIdx.x >> 5;
  const int ln = lane & 15, hi = lane >> 4;
  const int bh = blockIdx.y;
  const int qbase = (blockIdx.x * 4 + wave) * 16;

  const unsigned short* Qh = qB + (size_t)bh * Tz * Dz;
  const unsigned short* Kh = kB + (size_t)bh * Tz * Dz;
  const unsigned short* Vh = vtB + (size_t)bh * Dz * Tz;

  // Q^T as B-fragments (N=query, K=dh), resident for the whole row block
  const v16bf qf0 = ldfragB(Qh, Dz, qbase + ln, 0, hi);
  const v16bf qf1 = ldfragB(Qh, Dz, qbase + ln, 32, hi);

  v8f o[4] = {};                 // O^T accum: 4 dh-tiles of 16 x 16 queries
  float m = -3.0e38f, l = 0.0f;  // per-query running max / sum (lane = query)
  const int qi = qbase + ln;
  const int kend = qbase + 16;   // causal: keys < qbase+16 (masked within tile)

  for (int kb = 0; kb < kend; kb += 32) {
    // ---- S^T tile: 32 keys x 16 queries (2 C fragments) ----
    v8f s0 = {}, s1 = {};
    {
      v16bf ka0 = ldfragA(Kh, Dz, kb + ln,      0,  hi);
      v16bf ka1 = ldfragA(Kh, Dz, kb + ln,      32, hi);
      s0 = wmma_bf16(ka0, qf0, s0);
      s0 = wmma_bf16(ka1, qf1, s0);
      v16bf kc0 = ldfragA(Kh, Dz, kb + 16 + ln, 0,  hi);
      v16bf kc1 = ldfragA(Kh, Dz, kb + 16 + ln, 32, hi);
      s1 = wmma_bf16(kc0, qf0, s1);
      s1 = wmma_bf16(kc1, qf1, s1);
    }
    // ---- causal mask (only the diagonal-straddling tiles) ----
    if (kb + 31 > qbase) {
      for (int r = 0; r < 8; ++r) {
        const int key = kb + r + 8 * hi;
        if (key > qi)      s0[r] = -3.0e38f;
        if (key + 16 > qi) s1[r] = -3.0e38f;
      }
    }
    // ---- online softmax (exp2 domain; scale folded into Q) ----
    float tmax = -3.0e38f;
    for (int r = 0; r < 8; ++r) tmax = fmaxf(tmax, fmaxf(s0[r], s1[r]));
    tmax = fmaxf(tmax, __shfl_xor(tmax, 16, 32));
    const float mnew = fmaxf(m, tmax);
    const float corr = exp2f(m - mnew);
    float p0[8], p1[8], rsum = 0.0f;
    for (int r = 0; r < 8; ++r) {
      p0[r] = exp2f(s0[r] - mnew);
      p1[r] = exp2f(s1[r] - mnew);
      rsum += p0[r] + p1[r];
    }
    rsum += __shfl_xor(rsum, 16, 32);
    l = l * corr + rsum;
    m = mnew;
    for (int i = 0; i < 4; ++i)
      for (int r = 0; r < 8; ++r) o[i][r] *= corr;

    // ---- P^T C-layout -> B-fragment (one cross-half shuffle per value) ----
    Frag pf;
    for (int j = 0; j < 8; ++j) {
      const float own  = hi ? p1[j] : p0[j];
      const float send = hi ? p0[j] : p1[j];
      const float rec  = __shfl_xor(send, 16, 32);
      pf.us[j]     = f2bf(hi ? rec : own);   // keys [win+0 .. win+15]
      pf.us[8 + j] = f2bf(hi ? own : rec);
    }
    // ---- O^T += V^T * P^T (V^T rows are contiguous: V stored d-major) ----
    for (int i = 0; i < 4; ++i) {
      v16bf vf = ldfragA(Vh, Tz, i * 16 + ln, kb, hi);
      o[i] = wmma_bf16(vf, pf.bf, o[i]);
    }
  }

  // ---- normalize and store to attn buffer [b, t, h, d] (bf16) ----
  const float inv = 1.0f / l;
  const int h = bh & (Hz - 1), b = bh >> 4;
  const int t = qbase + ln;
  unsigned short* dst = oB + (((size_t)(b * Tz + t)) * Hz + h) * Dz;
  for (int i = 0; i < 4; ++i) {
    for (int r = 0; r < 8; r += 2) {
      union { unsigned short h2[2]; unsigned int u; } pk;
      pk.h2[0] = f2bf(o[i][r] * inv);
      pk.h2[1] = f2bf(o[i][r + 1] * inv);
      *(unsigned int*)(dst + i * 16 + 8 * hi + r) = pk.u;
    }
  }
}

// -------------------------------------------------------------------------
// Output projection: out = attn @ w_proj^T + b_proj  (fp32 output)
// -------------------------------------------------------------------------
__global__ void __launch_bounds__(256)
proj_kernel(const unsigned short* __restrict__ ab,
            const unsigned short* __restrict__ wb,
            const float* __restrict__ bias,
            float* __restrict__ out) {
  const int lane = threadIdx.x & 31, wave = threadIdx.x >> 5;
  const int ln = lane & 15, hi = lane >> 4;
  const int wm = wave >> 1, wn = wave & 1;
  const int mbase = blockIdx.y * 128 + wm * 32;
  const int nbase = blockIdx.x * 128 + wn * 64;

  v8f acc[2][4] = {};
  gemm_core(ab, wb, Ez, mbase, nbase, ln, hi, acc);

  for (int nt = 0; nt < 4; ++nt) {
    const int n = nbase + nt * 16 + ln;
    const float bv = bias[n];
    for (int mt = 0; mt < 2; ++mt) {
      for (int r = 0; r < 8; ++r) {
        const int mm = mbase + mt * 16 + r + 8 * hi;
        out[(size_t)mm * Ez + n] = acc[mt][nt][r] + bv;
      }
    }
  }
}

// -------------------------------------------------------------------------
extern "C" void kernel_launch(void* const* d_in, const int* in_sizes, int n_in,
                              void* d_out, int out_size, void* d_ws, size_t ws_size,
                              hipStream_t stream) {
  const float* x      = (const float*)d_in[0];
  const float* w_qkv  = (const float*)d_in[1];
  const float* b_qkv  = (const float*)d_in[2];
  const float* w_proj = (const float*)d_in[3];
  const float* b_proj = (const float*)d_in[4];
  float* out = (float*)d_out;

  char* ws = (char*)d_ws;
  auto alloc_bf = [&](size_t elems) {
    unsigned short* p = (unsigned short*)ws;
    ws += ((elems * 2 + 255) & ~(size_t)255);
    return p;
  };
  unsigned short* xb  = alloc_bf((size_t)Bz * Tz * Ez);      // x bf16
  unsigned short* wqb = alloc_bf((size_t)3 * Ez * Ez);       // w_qkv bf16
  unsigned short* wpb = alloc_bf((size_t)Ez * Ez);           // w_proj bf16
  unsigned short* qb  = alloc_bf((size_t)Bz * Hz * Tz * Dz); // Q  [b,h,t,d]
  unsigned short* kb  = alloc_bf((size_t)Bz * Hz * Tz * Dz); // K  [b,h,t,d]
  unsigned short* vtb = alloc_bf((size_t)Bz * Hz * Dz * Tz); // V^T[b,h,d,t]
  unsigned short* atb = alloc_bf((size_t)Bz * Tz * Ez);      // attn out [b,t,h,d]

  auto cvt = [&](const float* src, unsigned short* dst, int n) {
    cvt_kernel<<<dim3((unsigned)((n / 4 + 255) / 256)), dim3(256), 0, stream>>>(src, dst, n);
  };
  cvt(x,      xb,  Bz * Tz * Ez);
  cvt(w_qkv,  wqb, 3 * Ez * Ez);
  cvt(w_proj, wpb, Ez * Ez);

  qkv_kernel<<<dim3((3 * Ez) / 128, (Bz * Tz) / 128), dim3(256), 0, stream>>>(
      xb, wqb, b_qkv, qb, kb, vtb);

  attn_kernel<<<dim3(Tz / 64, Bz * Hz), dim3(128), 0, stream>>>(
      qb, kb, vtb, atb);

  proj_kernel<<<dim3(Ez / 128, (Bz * Tz) / 128), dim3(256), 0, stream>>>(
      atb, wpb, b_proj, out);
}
